// GraphConv__37194416783908
// MI455X (gfx1250) — compile-verified, matrix operand order checked by reading, not run on hardware
//
#include <hip/hip_runtime.h>
#include <hip/hip_bf16.h>

#define N_NODES 40000
#define N_EDGES 640000
#define DIN 128

typedef __attribute__((ext_vector_type(16))) __bf16 v16bf;
typedef __attribute__((ext_vector_type(8)))  float  v8f;

__device__ __forceinline__ unsigned int bf16_1(float f) {
    unsigned int u = __float_as_uint(f);
    u += 0x7FFFu + ((u >> 16) & 1u);   // round-to-nearest-even
    return u >> 16;
}
__device__ __forceinline__ unsigned int pack_bf16(float lo, float hi) {
    return bf16_1(lo) | (bf16_1(hi) << 16);
}

// ---------------------------------------------------------------- zero
__global__ void zero_kernel(float4* __restrict__ p, int n4) {
    int i = blockIdx.x * blockDim.x + threadIdx.x;
    if (i < n4) p[i] = make_float4(0.f, 0.f, 0.f, 0.f);
}

// ---------------------------------------------------------------- scatter (segment_sum)
// One wave per edge; src/dst are wave-uniform -> force SGPR so loads/atomics
// use the SADDR + lane-offset form. x (20.5MB) is L2-resident (192MB L2).
__global__ __launch_bounds__(256) void scatter_add_kernel(
    const float* __restrict__ x,
    const long long* __restrict__ src,
    const long long* __restrict__ dst,
    float* __restrict__ agg)
{
    int e = blockIdx.x * 8 + (threadIdx.x >> 5);
    int lane = threadIdx.x & 31;
    if (e >= N_EDGES) return;
    int s = __builtin_amdgcn_readfirstlane((int)src[e]);
    int d = __builtin_amdgcn_readfirstlane((int)dst[e]);
    float4 v = ((const float4*)(x + (long long)s * DIN))[lane];
    float* a = agg + (long long)d * DIN + lane * 4;
    atomicAdd(a + 0, v.x);
    atomicAdd(a + 1, v.y);
    atomicAdd(a + 2, v.z);
    atomicAdd(a + 3, v.w);
}

// ---------------------------------------------------------------- fused GEMM
// out[m][n] = relu( agg[m][:]@Wr[:,n] + h[m][:]@Ws[:,n] + br[n] )
// Block = 64-row M tile; wave w owns one 16-col N tile and 4 M sub-tiles
// (4 C fragments), so each staged B chunk is reused 4x. bf16 WMMA, f32 acc.
template <int DOUT>
__global__ __launch_bounds__(32 * (DOUT / 16)) void gemm_bias_relu_kernel(
    const float* __restrict__ agg, const float* __restrict__ h,
    const float* __restrict__ Wr, const float* __restrict__ br,
    const float* __restrict__ Ws, float* __restrict__ out)
{
    constexpr int NW = DOUT / 16;     // waves per block
    constexpr int T  = 32 * NW;       // threads per block
    constexpr int MT = 64;            // rows per block (4 x 16)

    // All LDS held as packed bf16 pairs (uint).
    __shared__ unsigned int sA1[MT * DIN / 2];   // agg tile   16KB
    __shared__ unsigned int sA2[MT * DIN / 2];   // h   tile   16KB
    __shared__ unsigned int sWr[DOUT * 16];      // Wr chunk, transposed [n][k/2]
    __shared__ unsigned int sWs[DOUT * 16];      // Ws chunk, transposed [n][k/2]

    const int tid  = threadIdx.x;
    const int wave = tid >> 5;
    const int lane = tid & 31;
    const int m0   = blockIdx.x * MT;

    // ---- stage A tiles once: float4 global loads -> packed bf16 -> b64 LDS stores
    for (int fid = tid; fid < MT * DIN / 4; fid += T) {
        int r  = fid >> 5;             // DIN/4 == 32 float4 per row
        int c4 = fid & 31;
        long long g = (long long)(m0 + r) * (DIN / 4) + c4;
        float4 v1 = ((const float4*)agg)[g];
        float4 v2 = ((const float4*)h)[g];
        uint2 p1 = make_uint2(pack_bf16(v1.x, v1.y), pack_bf16(v1.z, v1.w));
        uint2 p2 = make_uint2(pack_bf16(v2.x, v2.y), pack_bf16(v2.z, v2.w));
        ((uint2*)sA1)[r * 32 + c4] = p1;
        ((uint2*)sA2)[r * 32 + c4] = p2;
    }

    const int nloc = lane & 15;
    const int n    = wave * 16 + nloc;        // global output column
    const int kb   = (lane >> 4) * 8;         // K sub-offset per half-wave

    float bias = br[n];
    v8f c0, c1, c2, c3;
#pragma unroll
    for (int j = 0; j < 8; ++j) { c0[j] = bias; c1[j] = bias; c2[j] = bias; c3[j] = bias; }

    for (int kc = 0; kc < DIN / 32; ++kc) {
        __syncthreads();
        // ---- stage 32xDOUT weight chunks transposed & pair-packed:
        // sW[nn*16 + kp] = { W[(kc*32+2kp)*DOUT+nn], W[(kc*32+2kp+1)*DOUT+nn] }
        for (int pid = tid; pid < 16 * DOUT; pid += T) {
            int kp = pid / DOUT;               // 0..15 (pair index)
            int nn = pid % DOUT;               // coalesced global read over nn
            int gk = (kc * 32 + 2 * kp) * DOUT + nn;
            sWr[nn * 16 + kp] = pack_bf16(Wr[gk], Wr[gk + DOUT]);
            sWs[nn * 16 + kp] = pack_bf16(Ws[gk], Ws[gk + DOUT]);
        }
        __syncthreads();

        // Fragment K map (16-bit A 16x32 / B 32x16 layouts):
        //   VGPR j pairs: j<4 -> K = kb+2j ; j>=4 -> K = 16+kb+2(j-4)
        union { v16bf v; unsigned int u[8]; } b1, b2;
#pragma unroll
        for (int j = 0; j < 8; ++j) {
            int kl = (j < 4) ? (kb + 2 * j) : (16 + kb + 2 * (j - 4));
            b1.u[j] = sWr[n * 16 + (kl >> 1)];
            b2.u[j] = sWs[n * 16 + (kl >> 1)];
        }

#pragma unroll
        for (int mt = 0; mt < 4; ++mt) {
            const int m = mt * 16 + nloc;      // A row for this lane
            union { v16bf v; unsigned int u[8]; } a1, a2;
#pragma unroll
            for (int j = 0; j < 8; ++j) {
                int kl = (j < 4) ? (kb + 2 * j) : (16 + kb + 2 * (j - 4));
                int ki = (kc * 32 + kl) >> 1;
                a1.u[j] = sA1[m * (DIN / 2) + ki];
                a2.u[j] = sA2[m * (DIN / 2) + ki];
            }
            v8f* cf = (mt == 0) ? &c0 : (mt == 1) ? &c1 : (mt == 2) ? &c2 : &c3;
            *cf = __builtin_amdgcn_wmma_f32_16x16x32_bf16(false, a1.v, false, b1.v,
                                                          (short)0, *cf, false, false);
            *cf = __builtin_amdgcn_wmma_f32_16x16x32_bf16(false, a2.v, false, b2.v,
                                                          (short)0, *cf, false, false);
        }
    }

    // ---- ReLU + store. C/D layout: VGPR j -> row j + 8*(lane>=16), col = lane&15.
    const int mhi = (lane >> 4) * 8;
#pragma unroll
    for (int mt = 0; mt < 4; ++mt) {
        v8f* cf = (mt == 0) ? &c0 : (mt == 1) ? &c1 : (mt == 2) ? &c2 : &c3;
#pragma unroll
        for (int j = 0; j < 8; ++j) {
            float v = (*cf)[j];
            v = v > 0.f ? v : 0.f;
            out[(long long)(m0 + mt * 16 + mhi + j) * DOUT + n] = v;
        }
    }
}

// ---------------------------------------------------------------- log_softmax over 64 cols
__global__ __launch_bounds__(256) void log_softmax64_kernel(float* __restrict__ out, int nrows)
{
    int row  = blockIdx.x * 8 + (threadIdx.x >> 5);
    int lane = threadIdx.x & 31;
    if (row >= nrows) return;
    float2 v = ((const float2*)(out + (long long)row * 64))[lane];
    float m = fmaxf(v.x, v.y);
#pragma unroll
    for (int o = 16; o > 0; o >>= 1) m = fmaxf(m, __shfl_xor(m, o, 32));
    float e = expf(v.x - m) + expf(v.y - m);
#pragma unroll
    for (int o = 16; o > 0; o >>= 1) e += __shfl_xor(e, o, 32);
    float lse = m + logf(e);
    float2 r; r.x = v.x - lse; r.y = v.y - lse;
    ((float2*)(out + (long long)row * 64))[lane] = r;
}

// ---------------------------------------------------------------- launch
extern "C" void kernel_launch(void* const* d_in, const int* in_sizes, int n_in,
                              void* d_out, int out_size, void* d_ws, size_t ws_size,
                              hipStream_t stream) {
    const float*     x   = (const float*)d_in[0];
    const long long* ei  = (const long long*)d_in[1];   // int64, shape (2, E) row-major
    const long long* src = ei;
    const long long* dst = ei + N_EDGES;
    const float* Wr0 = (const float*)d_in[2];
    const float* br0 = (const float*)d_in[3];
    const float* Ws0 = (const float*)d_in[4];
    const float* Wr1 = (const float*)d_in[5];
    const float* br1 = (const float*)d_in[6];
    const float* Ws1 = (const float*)d_in[7];
    const float* Wr2 = (const float*)d_in[8];
    const float* br2 = (const float*)d_in[9];
    const float* Ws2 = (const float*)d_in[10];

    float* agg = (float*)d_ws;                          // 40000*128 f32
    float* h   = agg + (size_t)N_NODES * DIN;           // 40000*128 f32
    float* out = (float*)d_out;                         // 40000*64  f32

    const int n4     = N_NODES * DIN / 4;               // 1,280,000 float4
    const int zgrid  = (n4 + 255) / 256;
    const int sgrid  = N_EDGES / 8;                     // one wave per edge
    const int ggrid  = N_NODES / 64;                    // 625 64-row M tiles
    const int smgrid = N_NODES / 8;                     // one wave per row

    // ---- layer 0: agg = seg_sum(x); h = relu(agg@Wr0 + x@Ws0 + br0)
    zero_kernel<<<zgrid, 256, 0, stream>>>((float4*)agg, n4);
    scatter_add_kernel<<<sgrid, 256, 0, stream>>>(x, src, dst, agg);
    gemm_bias_relu_kernel<128><<<ggrid, 256, 0, stream>>>(agg, x, Wr0, br0, Ws0, h);

    // ---- layer 1 (in-place on h: block stages its 64 rows before writing them)
    zero_kernel<<<zgrid, 256, 0, stream>>>((float4*)agg, n4);
    scatter_add_kernel<<<sgrid, 256, 0, stream>>>(h, src, dst, agg);
    gemm_bias_relu_kernel<128><<<ggrid, 256, 0, stream>>>(agg, h, Wr1, br1, Ws1, h);

    // ---- layer 2 (DOUT = 64)
    zero_kernel<<<zgrid, 256, 0, stream>>>((float4*)agg, n4);
    scatter_add_kernel<<<sgrid, 256, 0, stream>>>(h, src, dst, agg);
    gemm_bias_relu_kernel<64><<<ggrid, 128, 0, stream>>>(agg, h, Wr2, br2, Ws2, out);

    // ---- log_softmax over 64 columns
    log_softmax64_kernel<<<smgrid, 256, 0, stream>>>(out, N_NODES);
}
